// Block_22651657519411
// MI455X (gfx1250) — matrix operand order repaired and checked by reading, not compiled, not run on hardware
//
#include <hip/hip_runtime.h>
#include <hip/hip_bf16.h>
#include <math.h>

// ---- problem constants (match reference) ------------------------------------
#define NN 4
#define CC 128
#define TT 128
#define VV 48
#define WW 3
#define UU 144          // W*V
#define SS 8
#define CDQ 16          // C/S
#define TE (TT + 2)     // temporally extended grid (pad rows at te=0, te=TE-1)

typedef __attribute__((ext_vector_type(16))) __bf16 bf16x16;
typedef __attribute__((ext_vector_type(8)))  __bf16 bf16x8;
typedef __attribute__((ext_vector_type(8)))  float  f32x8;

__device__ __forceinline__ f32x8 wmma_bf16(bf16x16 a, bf16x16 b, f32x8 c) {
  return __builtin_amdgcn_wmma_f32_16x16x32_bf16(false, a, false, b, (short)0, c,
                                                 false, false);
}

__device__ __forceinline__ bf16x16 combine8(bf16x8 lo, bf16x8 hi) {
  bf16x16 r;
#pragma unroll
  for (int i = 0; i < 8; ++i) { r[i] = lo[i]; r[8 + i] = hi[i]; }
  return r;
}

// ---- fragment loaders (contiguous-K lane order, not the ISA table order), all-contiguous sources -----
// A 16x32: lane(m,h): elem i -> K = k0 + 16h + i.  Contiguous-K row.
__device__ __forceinline__ bf16x16 frag_a_row(const __bf16* rowptr, int k0) {
  int h = (threadIdx.x & 31) >> 4;
  bf16x8 lo = *(const bf16x8*)(rowptr + k0 + 16 * h);
  bf16x8 hi = *(const bf16x8*)(rowptr + k0 + 16 * h + 8);
  return combine8(lo, hi);
}
// B 32x16: lane(nn,h): elem i -> K = k0 + 16h + i.  Contiguous-K column.
__device__ __forceinline__ bf16x16 frag_b_contig(const __bf16* base, int k0) {
  int h = (threadIdx.x & 31) >> 4;
  const __bf16* p = base + k0 + 16 * h;
  return combine8(*(const bf16x8*)p, *(const bf16x8*)(p + 8));
}
// Same, f32 source with convert (wo-accumulator tensor stays f32 for accuracy).
__device__ __forceinline__ bf16x16 frag_b_contig_f32(const float* base, int k0) {
  int h = (threadIdx.x & 31) >> 4;
  const float* p = base + k0 + 16 * h;
  bf16x16 r;
#pragma unroll
  for (int i = 0; i < 16; ++i) r[i] = (__bf16)p[i];
  return r;
}

// ---- weight pre-conversion (once, tiny) -------------------------------------
__global__ void cvt_kernel(const float* __restrict__ in, __bf16* __restrict__ out,
                           int n) {
  int i = blockIdx.x * 256 + threadIdx.x;
  if (i < n) out[i] = (__bf16)in[i];
}
// agg_w[(o*C+c)*W + j] -> awr[o*384 + j*128 + c]   (k = j*C + c contiguous)
__global__ void aggw_kernel(const float* __restrict__ aw, __bf16* __restrict__ out) {
  int i = blockIdx.x * 256 + threadIdx.x;
  if (i >= CC * CC * WW) return;
  int o = i / (CC * WW), r = i % (CC * WW);
  int j = r / CC, c = r % CC;
  out[i] = (__bf16)aw[(o * CC + c) * WW + j];
}

// ---- K1 / K6: channel LayerNorm -> channel-last bf16 ------------------------
// in_chlast=0: in[((n*C+c)*T+t)*V+v] (x).  in_chlast=1: in[((n*T+t)*V+v)*C+c].
__global__ void ln_kernel(const float* __restrict__ in, const float* __restrict__ g,
                          const float* __restrict__ beta, __bf16* __restrict__ out,
                          int pad, int in_chlast) {
  const int te_dim = pad ? TE : TT;
  int idx = blockIdx.x * blockDim.x + threadIdx.x;
  int total = NN * te_dim * VV;
  if (idx >= total) return;
  int v  = idx % VV;
  int te = (idx / VV) % te_dim;
  int n  = idx / (VV * te_dim);
  __bf16* op = out + ((size_t)(n * te_dim + te) * VV + v) * CC;
  if (pad && (te == 0 || te == te_dim - 1)) {
    for (int c = 0; c < CC; ++c) op[c] = (__bf16)beta[c];
    return;
  }
  int t = pad ? te - 1 : te;
  const float* ip;
  size_t stride;
  if (in_chlast) { ip = in + ((size_t)(n * TT + t) * VV + v) * CC; stride = 1; }
  else           { ip = in + ((size_t)n * CC * TT + t) * VV + v;   stride = (size_t)TT * VV; }
  float sum = 0.f, sq = 0.f;
  for (int c = 0; c < CC; ++c) {
    float xv = ip[c * stride];
    sum += xv; sq += xv * xv;
  }
  float mu  = sum * (1.f / CC);
  float var = fmaxf(sq * (1.f / CC) - mu * mu, 0.f);
  float rs  = rsqrtf(var + 1e-5f);
  for (int c = 0; c < CC; ++c) {
    float xv = ip[c * stride];
    op[c] = (__bf16)((xv - mu) * rs * g[c] + beta[c]);
  }
}

// ---- K2/K7/K8: 128x128 bf16 weight GEMM, 64(out-ch) x 16(pos) per wave ------
// X channel-last bf16. MODE 0: bf16 channel-first out (v).  MODE 3: bf16
// channel-last (q,k,FFN-hidden chain).  MODE 1: GELU -> bf16 channel-last.
// MODE 2: +resid(channel-last f32) -> f32 channel-first (final output).
template <int MODE>
__global__ void gemm128_kernel(const __bf16* __restrict__ Wb, const float* __restrict__ bias,
                               const __bf16* __restrict__ X, int ncols,
                               __bf16* __restrict__ outb, float* __restrict__ outf,
                               const float* __restrict__ resid) {
  const int lane = threadIdx.x & 31;
  const int m = lane & 15, h = lane >> 4;
  const int ct = blockIdx.x, ot = blockIdx.y, n = blockIdx.z;
  const int col = ct * 16 + m;
  const __bf16* colbase = X + ((size_t)n * ncols + col) * CC;
  const __bf16* wrow[4];
#pragma unroll
  for (int p = 0; p < 4; ++p) wrow[p] = Wb + (size_t)(ot * 64 + p * 16 + m) * CC;
  f32x8 acc[4] = {{}, {}, {}, {}};
#pragma unroll
  for (int k0 = 0; k0 < CC; k0 += 32) {
    bf16x16 bfr = frag_b_contig(colbase, k0);
#pragma unroll
    for (int p = 0; p < 4; ++p)
      acc[p] = wmma_bf16(frag_a_row(wrow[p], k0), bfr, acc[p]);
  }
#pragma unroll
  for (int p = 0; p < 4; ++p) {
    int o0 = ot * 64 + p * 16 + 8 * h;
    if (MODE == 3 || MODE == 1) {
      bf16x8 vs;
#pragma unroll
      for (int r = 0; r < 8; ++r) {
        float val = acc[p][r] + bias[o0 + r];
        if (MODE == 1) val = 0.5f * val * (1.f + erff(val * 0.70710678118f));
        vs[r] = (__bf16)val;
      }
      *(bf16x8*)(outb + ((size_t)n * ncols + col) * CC + o0) = vs;
    } else if (MODE == 0) {
#pragma unroll
      for (int r = 0; r < 8; ++r)
        outb[(size_t)(n * CC + o0 + r) * ncols + col] = (__bf16)(acc[p][r] + bias[o0 + r]);
    } else {  // MODE 2
      const float* rp = resid + ((size_t)n * ncols + col) * CC + o0;
#pragma unroll
      for (int r = 0; r < 8; ++r)
        outf[(size_t)(n * CC + o0 + r) * ncols + col] = acc[p][r] + bias[o0 + r] + rp[r];
    }
  }
}

// ---- K_attn: attnT[n,s,u',u] = tanh(q^T k/(CD*T))*alpha_s + ga;  48x48/wave -
// q/k channel-last on extended grid: q[((n*TE+te)*V+v)*C + s*16 + c].
__global__ void attn_kernel(const __bf16* __restrict__ q, const __bf16* __restrict__ kx,
                            const float* __restrict__ alphas, const float* __restrict__ ga,
                            __bf16* __restrict__ attnT) {
  const int lane = threadIdx.x & 31;
  const int m = lane & 15, h = lane >> 4;
  const int u2t = blockIdx.x, ut = blockIdx.y;
  const int n = blockIdx.z >> 3, s = blockIdx.z & 7;
  const size_t PS = (size_t)VV * CC;                 // t -> t+1 position stride
  // hoisted window decode (u = j*V + v -> te = t + j)
  const __bf16* qbase[3];   // A rows: + 8h channel sub-offset
  const __bf16* kbase[3];   // B cols
#pragma unroll
  for (int p = 0; p < 3; ++p) {
    int u  = ut * 48 + p * 16 + m;
    qbase[p] = q + ((size_t)(n * TE + u / VV) * VV + u % VV) * CC + s * CDQ + 8 * h;
    int u2 = u2t * 48 + p * 16 + m;
    kbase[p] = kx + ((size_t)(n * TE + u2 / VV) * VV + u2 % VV) * CC + s * CDQ;
  }
  f32x8 acc[3][3] = {};
  for (int k0 = 0; k0 < CDQ * TT; k0 += 32) {
    int t0 = k0 >> 4;
    if (k0 + 32 < CDQ * TT) {
      __builtin_prefetch(qbase[0] + (size_t)(t0 + 2) * PS, 0, 1);
      __builtin_prefetch(kbase[0] + (size_t)(t0 + 2) * PS, 0, 1);
    }
    bf16x16 a[3], b[3];
#pragma unroll
    for (int p = 0; p < 3; ++p) {
      // A elem i: c = 8h+(i&7); t = t0 (+1 for i>=8): two 16B loads
      const __bf16* pa = qbase[p] + (size_t)t0 * PS;
      a[p] = combine8(*(const bf16x8*)pa, *(const bf16x8*)(pa + PS));
      // B elem i: c = 8h+(i&7); t = t0 (+1 for i>=8): two 16B loads
      const __bf16* pb = kbase[p] + (size_t)t0 * PS + 8 * h;
      b[p] = combine8(*(const bf16x8*)pb, *(const bf16x8*)(pb + PS));
    }
#pragma unroll
    for (int pi = 0; pi < 3; ++pi)
#pragma unroll
      for (int pj = 0; pj < 3; ++pj)
        acc[pi][pj] = wmma_bf16(a[pi], b[pj], acc[pi][pj]);
  }
  const float alpha = alphas[s];
  __bf16* abase = attnT + (size_t)(n * SS + s) * UU * UU;
#pragma unroll
  for (int pj = 0; pj < 3; ++pj) {
    int u2 = u2t * 48 + pj * 16 + m;
#pragma unroll
    for (int pi = 0; pi < 3; ++pi) {
      int u0 = ut * 48 + pi * 16 + 8 * h;
      bf16x8 vs;
#pragma unroll
      for (int r = 0; r < 8; ++r) {
        int u = u0 + r;
        vs[r] = (__bf16)(tanhf(acc[pi][pj][r] * (1.f / (CDQ * TT))) * alpha
                         + ga[u * UU + u2]);
      }
      *(bf16x8*)(abase + (size_t)u2 * UU + u0) = vs;   // transposed store
    }
  }
}

// ---- K3: m2[(t,u'),c] = sum_u v_win[(c,t),u] * attnT[u',u];  64x16/wave -----
// v channel-first: window gather contiguous in u ((t+j)*V+v = t*V+u).
__global__ void av_kernel(const __bf16* __restrict__ vb, const __bf16* __restrict__ attnT,
                          __bf16* __restrict__ m2, int n, int s) {
  const int lane = threadIdx.x & 31;
  const int mm = lane & 15, h = lane >> 4;
  const int ct = blockIdx.x;    // 0..8   over u'
  const int rt = blockIdx.y;    // 0..255 over (c*T+t)/64
  const __bf16* vrow[4];
#pragma unroll
  for (int p = 0; p < 4; ++p) {
    int row = rt * 64 + p * 16 + mm;
    int c = row >> 7, t = row & 127;
    vrow[p] = vb + ((size_t)(n * CC + c) * TE + t) * VV;
  }
  const int u2col = ct * 16 + mm;
  const __bf16* colT = attnT + ((size_t)(n * SS + s) * UU + u2col) * UU;
  f32x8 acc[4] = {{}, {}, {}, {}};
  for (int k0 = 0; k0 < UU; k0 += 32) {            // 5 steps, last partial
    bf16x16 bfr;
    {
      int kb = k0 + 8 * h;                         // lo always < UU; hi may be out
      bf16x8 blo = *(const bf16x8*)(colT + kb);
      bf16x8 bhi;
      if (kb + 16 < UU) bhi = *(const bf16x8*)(colT + kb + 16);
      else { bf16x8 z = {}; bhi = z; }
      bfr = combine8(blo, bhi);
    }
#pragma unroll
    for (int p = 0; p < 4; ++p) {
      int kA = k0 + 8 * h;        // always < UU (max 136)
      int kB = k0 + 16 + 8 * h;   // out of range only on last step
      bf16x8 lo = *(const bf16x8*)(vrow[p] + kA);
      bf16x8 hi;
      if (kB < UU) hi = *(const bf16x8*)(vrow[p] + kB);
      else { bf16x8 z = {}; hi = z; }
      acc[p] = wmma_bf16(combine8(lo, hi), bfr, acc[p]);
    }
  }
#pragma unroll
  for (int p = 0; p < 4; ++p)
#pragma unroll
    for (int r = 0; r < 8; ++r) {
      int row = rt * 64 + p * 16 + r + 8 * h;
      int c = row >> 7, t = row & 127;
      m2[((size_t)t * UU + u2col) * CC + c] = (__bf16)acc[p][r];
    }
}

// ---- K4: att2[(t,u),o] (+)= wo_s @ m2 ; channel-last f32 accumulator --------
__global__ void wo_kernel(const __bf16* __restrict__ wob, const float* __restrict__ bo,
                          const __bf16* __restrict__ m2, float* __restrict__ att2,
                          int n, int s, int first) {
  const int lane = threadIdx.x & 31;
  const int mm = lane & 15, h = lane >> 4;
  const int ct = blockIdx.x;      // 0..1151 over (t,u)
  const int ot = blockIdx.y;      // 0..1
  const int NCOL = TT * UU;       // 18432
  const int col = ct * 16 + mm;
  const __bf16* wrow[4];
#pragma unroll
  for (int p = 0; p < 4; ++p)
    wrow[p] = wob + (size_t)(ot * 64 + p * 16 + mm) * (SS * CC) + s * CC;
  const __bf16* colbase = m2 + (size_t)col * CC;
  f32x8 acc[4] = {{}, {}, {}, {}};
#pragma unroll
  for (int k0 = 0; k0 < CC; k0 += 32) {
    bf16x16 bfr = frag_b_contig(colbase, k0);
#pragma unroll
    for (int p = 0; p < 4; ++p)
      acc[p] = wmma_bf16(frag_a_row(wrow[p], k0), bfr, acc[p]);
  }
#pragma unroll
  for (int p = 0; p < 4; ++p) {
    int o0 = ot * 64 + p * 16 + 8 * h;
    float* ab = att2 + ((size_t)n * NCOL + col) * CC + o0;   // contiguous 8
#pragma unroll
    for (int r = 0; r < 8; ++r) {
      float base = first ? bo[o0 + r] : ab[r];
      ab[r] = base + acc[p][r];
    }
  }
}

// ---- K5: aggregate over (j,c) + residual -> y1 channel-last f32 -------------
// k = j*128 + c ; awr[o][j][c] contiguous; att2[(t, j*V+vv)][c] contiguous in c.
__global__ void agg_kernel(const __bf16* __restrict__ awr, const float* __restrict__ ab,
                           const float* __restrict__ att2, const float* __restrict__ x,
                           float* __restrict__ y1) {
  const int lane = threadIdx.x & 31;
  const int mm = lane & 15, h = lane >> 4;
  const int ct = blockIdx.x;   // 0..383 over T*V
  const int ot = blockIdx.y;   // 0..1
  const int n  = blockIdx.z;
  const int col = ct * 16 + mm;
  const int t = col / VV, vv = col % VV;             // hoisted once
  const float* bbase = att2 + ((size_t)n * TT * UU + (size_t)t * UU + vv) * CC;
  const __bf16* awrow[4];
#pragma unroll
  for (int p = 0; p < 4; ++p)
    awrow[p] = awr + (size_t)(ot * 64 + p * 16 + mm) * (CC * WW);
  f32x8 acc[4] = {{}, {}, {}, {}};
#pragma unroll
  for (int k0 = 0; k0 < CC * WW; k0 += 32) {
    bf16x16 bfr;
    {
      int k = k0 + 16 * h;               // chunk never straddles j (128 | 16)
      int j = k >> 7, c0 = k & 127;
      bfr = frag_b_contig_f32(bbase + (size_t)j * VV * CC + c0 - (size_t)(16 * h), k0 - k0);
      // note: frag_b_contig_f32 re-adds k0+16h; cancel it by passing base-16h, k0=0
      const float* p = bbase + (size_t)j * VV * CC + c0;
#pragma unroll
      for (int i = 0; i < 16; ++i) bfr[i] = (__bf16)p[i];
    }
#pragma unroll
    for (int p = 0; p < 4; ++p)
      acc[p] = wmma_bf16(frag_a_row(awrow[p], k0), bfr, acc[p]);
  }
#pragma unroll
  for (int p = 0; p < 4; ++p) {
    int o0 = ot * 64 + p * 16 + 8 * h;
    float* yp = y1 + ((size_t)n * TT * VV + col) * CC + o0;  // contiguous 8
#pragma unroll
    for (int r = 0; r < 8; ++r) {
      int o = o0 + r;
      float xv = x[((size_t)(n * CC + o) * TT + t) * VV + vv];
      yp[r] = acc[p][r] + ab[o] + xv;
    }
  }
}

// ---- launcher ---------------------------------------------------------------
extern "C" void kernel_launch(void* const* d_in, const int* in_sizes, int n_in,
                              void* d_out, int out_size, void* d_ws, size_t ws_size,
                              hipStream_t stream) {
  (void)in_sizes; (void)n_in; (void)out_size; (void)ws_size;
  const float* x      = (const float*)d_in[0];
  const float* wq     = (const float*)d_in[1];
  const float* bq     = (const float*)d_in[2];
  const float* wk     = (const float*)d_in[3];
  const float* bk     = (const float*)d_in[4];
  const float* wv     = (const float*)d_in[5];
  const float* bv     = (const float*)d_in[6];
  const float* wo     = (const float*)d_in[7];
  const float* bo     = (const float*)d_in[8];
  const float* alphas = (const float*)d_in[9];
  const float* ga     = (const float*)d_in[10];
  const float* agg_w  = (const float*)d_in[11];
  const float* agg_b  = (const float*)d_in[12];
  const float* ln1_g  = (const float*)d_in[13];
  const float* ln1_b  = (const float*)d_in[14];
  const float* ffn_g  = (const float*)d_in[15];
  const float* ffn_b  = (const float*)d_in[16];
  const float* w1     = (const float*)d_in[17];
  const float* b1     = (const float*)d_in[18];
  const float* w2     = (const float*)d_in[19];
  const float* b2     = (const float*)d_in[20];
  float* out = (float*)d_out;

  // workspace carve-out (~96 MB total)
  char* wsp = (char*)d_ws;
  size_t off = 0;
  auto carve = [&](size_t bytes) -> void* {
    off = (off + 255) & ~(size_t)255;
    void* p = wsp + off;
    off += bytes;
    return p;
  };
  const size_t ext_elems = (size_t)NN * CC * TE * VV;             // 3,194,880
  const size_t tv_elems  = (size_t)NN * CC * TT * VV;             // 3,145,728
  __bf16* xe  = (__bf16*)carve(ext_elems * 2);                    // channel-last
  __bf16* qcl = (__bf16*)carve(ext_elems * 2);                    // channel-last
  __bf16* kcl = (__bf16*)carve(ext_elems * 2);                    // channel-last
  __bf16* vcf = (__bf16*)carve(ext_elems * 2);                    // channel-first
  __bf16* atnT = (__bf16*)carve((size_t)NN * SS * UU * UU * 2);   // transposed
  __bf16* m2  = (__bf16*)carve((size_t)CC * TT * UU * 2);         // channel-last
  float*  att2 = (float*)carve((size_t)NN * CC * TT * UU * 4);    // channel-last
  float*  y1  = (float*)carve(tv_elems * 4);                      // channel-last
  __bf16* yn  = (__bf16*)carve(tv_elems * 2);                     // channel-last
  __bf16* hb  = (__bf16*)carve(tv_elems * 2);                     // channel-last
  __bf16* wqb = (__bf16*)carve((size_t)CC * CC * 2);
  __bf16* wkb = (__bf16*)carve((size_t)CC * CC * 2);
  __bf16* wvb = (__bf16*)carve((size_t)CC * CC * 2);
  __bf16* w1b = (__bf16*)carve((size_t)CC * CC * 2);
  __bf16* w2b = (__bf16*)carve((size_t)CC * CC * 2);
  __bf16* wob = (__bf16*)carve((size_t)CC * SS * CC * 2);
  __bf16* awr = (__bf16*)carve((size_t)CC * CC * WW * 2);

  const int EXTCOLS = TE * VV;  // 6240 (div by 16)
  const int TVCOLS  = TT * VV;  // 6144

  // K0: weight conversion / reorder (tiny, once per call)
  cvt_kernel<<<(CC * CC + 255) / 256, 256, 0, stream>>>(wq, wqb, CC * CC);
  cvt_kernel<<<(CC * CC + 255) / 256, 256, 0, stream>>>(wk, wkb, CC * CC);
  cvt_kernel<<<(CC * CC + 255) / 256, 256, 0, stream>>>(wv, wvb, CC * CC);
  cvt_kernel<<<(CC * CC + 255) / 256, 256, 0, stream>>>(w1, w1b, CC * CC);
  cvt_kernel<<<(CC * CC + 255) / 256, 256, 0, stream>>>(w2, w2b, CC * CC);
  cvt_kernel<<<(CC * SS * CC + 255) / 256, 256, 0, stream>>>(wo, wob, CC * SS * CC);
  aggw_kernel<<<(CC * CC * WW + 255) / 256, 256, 0, stream>>>(agg_w, awr);

  // K1: LayerNorm onto extended grid, channel-last (pad rows = ln1_b)
  {
    int total = NN * TE * VV;
    ln_kernel<<<(total + 255) / 256, 256, 0, stream>>>(x, ln1_g, ln1_b, xe, 1, 0);
  }
  // K2: q/k (channel-last) and v (channel-first) 1x1 convs
  dim3 gqkv(EXTCOLS / 16, CC / 64, NN);
  gemm128_kernel<3><<<gqkv, 32, 0, stream>>>(wqb, bq, xe, EXTCOLS, qcl, nullptr, nullptr);
  gemm128_kernel<3><<<gqkv, 32, 0, stream>>>(wkb, bk, xe, EXTCOLS, kcl, nullptr, nullptr);
  gemm128_kernel<0><<<gqkv, 32, 0, stream>>>(wvb, bv, xe, EXTCOLS, vcf, nullptr, nullptr);

  // K_attn: 48x48 macro-tile per wave, transposed bf16 store
  attn_kernel<<<dim3(UU / 48, UU / 48, NN * SS), 32, 0, stream>>>(qcl, kcl, alphas, ga, atnT);

  // K3+K4 per (n,s): m2 = V_win @ attn_s ; att2[n] (+)= wo_s @ m2
  for (int n = 0; n < NN; ++n) {
    for (int s = 0; s < SS; ++s) {
      av_kernel<<<dim3(UU / 16, (CC * TT) / 64), 32, 0, stream>>>(vcf, atnT, m2, n, s);
      wo_kernel<<<dim3((TT * UU) / 16, CC / 64), 32, 0, stream>>>(wob, bo, m2, att2, n, s,
                                                                  s == 0 ? 1 : 0);
    }
  }

  // K5: aggregate over window + residual with x -> y1 (channel-last)
  agg_kernel<<<dim3(TVCOLS / 16, CC / 64, NN), 32, 0, stream>>>(awr, agg_b, att2, x, y1);

  // K6: FFN LayerNorm (channel-last in and out)
  {
    int total = NN * TT * VV;
    ln_kernel<<<(total + 255) / 256, 256, 0, stream>>>(y1, ffn_g, ffn_b, yn, 0, 1);
  }
  // K7: FFN w1 + exact GELU (channel-last)
  dim3 gffn(TVCOLS / 16, CC / 64, NN);
  gemm128_kernel<1><<<gffn, 32, 0, stream>>>(w1b, b1, yn, TVCOLS, hb, nullptr, nullptr);
  // K8: FFN w2 + residual (y1) -> d_out (f32, reference NCTV layout)
  gemm128_kernel<2><<<gffn, 32, 0, stream>>>(w2b, b2, hb, TVCOLS, nullptr, out, y1);
}
